// MultiScaleGraphConv_76184129897201
// MI455X (gfx1250) — matrix operand order repaired and checked by reading, not compiled, not run on hardware
//
#include <hip/hip_runtime.h>
#include <hip/hip_bf16.h>
#include <math.h>

// ---------------------------------------------------------------------------
// MI455X / gfx1250: wave32, WMMA 16x16x32 bf16 (f32 accum), async LDS copy.
// Pipeline:
//   k_zero        : clear stat accumulators
//   k_prep_w      : W -> bf16 W1, (W2-W1); Wf -> bf16
//   k_transpose   : x (B,C,N) -> xtb (B*N, C) bf16 + squared norms sq
//   k_gram        : D[b,i,j] = sq_i + sq_j - 2 * <x_i,x_j>
//                   128x128 block tile; B panel staged in LDS via
//                   GLOBAL_LOAD_ASYNC_TO_LDS_B128; 32 WMMA per wave
//   k_topk        : per row, sorted top-32 nearest indices
//   k_feat_gemm   : y1 = X*W1^T ; z = X*(W2-W1)^T + bias        (WMMA bf16)
//   k_edge_stats  : channel sum / sumsq of h = y1[j] + z[i] over (B,N,k)
//   k_edge_coef   : BN coefficients per scale/channel
//   k_maxpool     : max_k leakyrelu(BN(h)) -> featb (B*N, 384) bf16
//   k_final_gemm  : y = feat * Wf^T + bf                        (WMMA bf16)
//   k_final_stats : channel sum / sumsq of y over (B,N)
//   k_final_out   : BN + exact GELU -> d_out (B, 384, N)
// ---------------------------------------------------------------------------

typedef __attribute__((ext_vector_type(16))) __bf16 v16bf;
typedef __attribute__((ext_vector_type(8)))  __bf16 v8bf;
typedef __attribute__((ext_vector_type(8)))  float  v8f;

#define PB   8
#define PC   128
#define PN   2048
#define BN   (PB * PN)       // 16384 points total
#define OUTC 384
#define OPER 128
#define LDB  136             // padded LDS row stride (bf16 elems): 68 DW -> 4-bank skew

union ABf { v16bf v; v8bf h[2]; };

// A tile 16x32 bf16 (M x K), source row-major [m][k], row stride ld.
// Lane L: row = L&15 ; K = k0 + (L>>4)*16 .. +15 contiguous.
__device__ __forceinline__ v16bf load_a_tile(const __bf16* __restrict__ base,
                                             int ld, int row0, int k0, int lane) {
  const __bf16* p = base + (size_t)(row0 + (lane & 15)) * ld + k0 + ((lane >> 4) << 4);
  ABf a;
  a.h[0] = *(const v8bf*)(p);
  a.h[1] = *(const v8bf*)(p + 8);
  return a.v;
}

// B tile 32x16 bf16 (K x N) taken from a row-major [n][k] matrix (i.e. B^T),
// row stride ld. Lane L: col n = L&15 ; K = k0 + (L>>4)*16 .. +15 contiguous.
__device__ __forceinline__ v16bf load_bt_tile(const __bf16* __restrict__ base,
                                              int ld, int col0, int k0, int lane) {
  const __bf16* p = base + (size_t)(col0 + (lane & 15)) * ld + k0 + ((lane >> 4) << 4);
  return *(const v16bf*)(p);
}

// Same B-tile pattern but from the padded LDS panel (two 16B halves).
__device__ __forceinline__ v16bf load_bt_lds(const __bf16* sm, int col0, int k0,
                                             int lane) {
  const __bf16* p = sm + (col0 + (lane & 15)) * LDB + k0 + ((lane >> 4) << 4);
  ABf b;
  b.h[0] = *(const v8bf*)(p);
  b.h[1] = *(const v8bf*)(p + 8);
  return b.v;
}

__device__ __forceinline__ v8f wmma_bf16(v16bf a, v16bf b, v8f c) {
  // (neg_a, A, neg_b, B, c_mod, C, reuse_a, reuse_b)
  return __builtin_amdgcn_wmma_f32_16x16x32_bf16(false, a, false, b, (short)0, c,
                                                 false, false);
}

// Async global -> LDS 16-byte copy (ASYNCcnt tracked).  VDST = LDS byte
// offset, VADDR = 64-bit global address (ISA 10.x / 15.18, opcode 98).
__device__ __forceinline__ void async_copy_b128(unsigned lds_off,
                                                const __bf16* gaddr) {
  asm volatile("global_load_async_to_lds_b128 %0, %1, off"
               :: "v"(lds_off), "v"(gaddr) : "memory");
}
__device__ __forceinline__ void wait_async0() {
  asm volatile("s_wait_asynccnt 0x0" ::: "memory");
}

// --------------------------------------------------------------------------

__global__ void k_zero(float* __restrict__ stats, float* __restrict__ fstats) {
  int t = blockIdx.x * blockDim.x + threadIdx.x;
  if (t < 768) stats[t]  = 0.0f;
  if (t < 768) fstats[t] = 0.0f;
}

__global__ void k_prep_w(const float* __restrict__ W0, const float* __restrict__ W1,
                         const float* __restrict__ W2, const float* __restrict__ Wf,
                         __bf16* __restrict__ w1b, __bf16* __restrict__ wdb,
                         __bf16* __restrict__ wfb) {
  int t = blockIdx.x * blockDim.x + threadIdx.x;
  if (t < 3 * OPER * PC) {
    int s = t / (OPER * PC);
    int r = t - s * (OPER * PC);
    int o = r / PC, k = r - o * PC;
    const float* W = (s == 0) ? W0 : ((s == 1) ? W1 : W2);
    float w1 = W[o * (2 * PC) + k];
    float w2 = W[o * (2 * PC) + PC + k];
    w1b[t] = (__bf16)w1;
    wdb[t] = (__bf16)(w2 - w1);
  }
  if (t < OUTC * OUTC) wfb[t] = (__bf16)Wf[t];
}

__global__ void k_transpose(const float* __restrict__ x, __bf16* __restrict__ xtb,
                            float* __restrict__ sq) {
  int p = blockIdx.x;            // 0 .. BN-1
  int b = p >> 11, n = p & (PN - 1);
  int c = threadIdx.x;           // 0 .. 127
  float v = x[((size_t)b * PC + c) * PN + n];
  xtb[(size_t)p * PC + c] = (__bf16)v;
  __shared__ float red[PC];
  red[c] = v * v;
  __syncthreads();
  for (int s = PC / 2; s > 0; s >>= 1) {
    if (c < s) red[c] += red[c + s];
    __syncthreads();
  }
  if (c == 0) sq[p] = red[0];
}

// grid (PN/128, PN/128, PB), block 256 (8 waves).
// Block computes a 128x128 D tile.  The 128x128 bf16 B panel (j rows) is
// staged once in LDS (async copies); each wave keeps its 16x128 A panel in
// VGPRs and sweeps 8 j-tiles x 4 k-chunks = 32 WMMAs.
__global__ void __launch_bounds__(256)
k_gram(const __bf16* __restrict__ xtb, const float* __restrict__ sq,
       float* __restrict__ D) {
  __shared__ __align__(32) __bf16 smem[128 * LDB];   // ~34 KB
  const int lane = threadIdx.x & 31;
  const int wv   = threadIdx.x >> 5;
  const int b    = blockIdx.z;
  const int i0   = blockIdx.y * 128 + wv * 16;
  const int j0   = blockIdx.x * 128;
  const __bf16* Xb = xtb + (size_t)b * PN * PC;

  // ---- async fill of the B panel: 128 rows x 128 bf16 (16B x 16 per row)
  const unsigned lds_base = (unsigned)(size_t)(&smem[0]);
  const int t = threadIdx.x;
#pragma unroll
  for (int i = 0; i < 8; ++i) {
    int id = t + i * 256;              // 0 .. 2047
    int r  = id >> 4;                  // row 0..127
    int c  = id & 15;                  // 16B chunk 0..15
    const __bf16* g = Xb + (size_t)(j0 + r) * PC + c * 8;
    async_copy_b128(lds_base + (unsigned)(r * LDB + c * 8) * 2, g);
  }

  // ---- preload the wave's A panel (16 rows x 128 K) into registers
  v16bf a0 = load_a_tile(Xb, PC, i0, 0,  lane);
  v16bf a1 = load_a_tile(Xb, PC, i0, 32, lane);
  v16bf a2 = load_a_tile(Xb, PC, i0, 64, lane);
  v16bf a3 = load_a_tile(Xb, PC, i0, 96, lane);

  const float* sqb = sq + (size_t)b * PN;
  float si[8];
#pragma unroll
  for (int r = 0; r < 8; ++r) si[r] = sqb[i0 + r + ((lane >> 4) << 3)];

  wait_async0();
  __syncthreads();

  float* Drow = D + ((size_t)b * PN) * PN;
#pragma unroll
  for (int jt = 0; jt < 8; ++jt) {
    v8f acc = {};
    acc = wmma_bf16(a0, load_bt_lds(smem, jt * 16, 0,  lane), acc);
    acc = wmma_bf16(a1, load_bt_lds(smem, jt * 16, 32, lane), acc);
    acc = wmma_bf16(a2, load_bt_lds(smem, jt * 16, 64, lane), acc);
    acc = wmma_bf16(a3, load_bt_lds(smem, jt * 16, 96, lane), acc);
    const int jcol = j0 + jt * 16 + (lane & 15);
    const float sj = sqb[jcol];
#pragma unroll
    for (int r = 0; r < 8; ++r) {
      int m = r + ((lane >> 4) << 3);
      Drow[(size_t)(i0 + m) * PN + jcol] = si[r] + sj - 2.0f * acc[r];
    }
  }
}

// one thread per row: sorted (ascending distance) top-32 indices
__global__ void k_topk(const float* __restrict__ D, int* __restrict__ idx) {
  int row = blockIdx.x * blockDim.x + threadIdx.x;   // 0..BN-1
  const float* d = D + (size_t)row * PN;
  float best[32];
  int   bidx[32];
#pragma unroll
  for (int i = 0; i < 32; ++i) { best[i] = 3.4e38f; bidx[i] = 0; }
  for (int j = 0; j < PN; ++j) {
    float v = d[j];
    if (v < best[31]) {
      int p = 31;
      while (p > 0 && best[p - 1] > v) {
        best[p] = best[p - 1]; bidx[p] = bidx[p - 1]; --p;
      }
      best[p] = v; bidx[p] = j;
    }
  }
  int* o = idx + (size_t)row * 32;
#pragma unroll
  for (int i = 0; i < 32; ++i) o[i] = bidx[i];
}

// grid (BN/128, OPER/16, 3), block 256. Wave: 16 rows x 16 out-channels,
// two accumulators (y1 and z) sharing the A loads.
__global__ void __launch_bounds__(256)
k_feat_gemm(const __bf16* __restrict__ xtb, const __bf16* __restrict__ w1b,
            const __bf16* __restrict__ wdb, const float* __restrict__ b0,
            const float* __restrict__ b1, const float* __restrict__ b2,
            float* __restrict__ y1, float* __restrict__ z) {
  const int lane = threadIdx.x & 31;
  const int wv   = threadIdx.x >> 5;
  const int s    = blockIdx.z;
  const int row0 = blockIdx.x * 128 + wv * 16;
  const int o0   = blockIdx.y * 16;
  const __bf16* W1 = w1b + (size_t)s * OPER * PC;
  const __bf16* Wd = wdb + (size_t)s * OPER * PC;
  const float*  bb = (s == 0) ? b0 : ((s == 1) ? b1 : b2);

  v8f acc1 = {}, acc2 = {};
#pragma unroll
  for (int kc = 0; kc < PC / 32; ++kc) {
    v16bf a   = load_a_tile(xtb, PC, row0, kc * 32, lane);
    v16bf bm1 = load_bt_tile(W1, PC, o0, kc * 32, lane);
    v16bf bm2 = load_bt_tile(Wd, PC, o0, kc * 32, lane);
    acc1 = wmma_bf16(a, bm1, acc1);
    acc2 = wmma_bf16(a, bm2, acc2);
  }

  const int   col  = o0 + (lane & 15);
  const float bias = bb[col];
  float* y1s = y1 + (size_t)s * BN * OPER;
  float* zs  = z  + (size_t)s * BN * OPER;
#pragma unroll
  for (int r = 0; r < 8; ++r) {
    int m = r + ((lane >> 4) << 3);
    size_t p = (size_t)(row0 + m);
    y1s[p * OPER + col] = acc1[r];
    zs [p * OPER + col] = acc2[r] + bias;
  }
}

// grid (BN/16, 3), block 128 (thread = channel). Accumulate sum/sumsq of
// h = y1[j] + z[i] over 16 points x k edges, one atomicAdd pair per thread.
__global__ void k_edge_stats(const float* __restrict__ y1, const float* __restrict__ z,
                             const int* __restrict__ idx, float* __restrict__ stats) {
  const int s = blockIdx.y;
  const int k = 8 << s;
  const int o = threadIdx.x;
  const float* y1s = y1 + (size_t)s * BN * OPER;
  const float* zs  = z  + (size_t)s * BN * OPER;
  int p0 = blockIdx.x * 16;
  float sum = 0.0f, sumsq = 0.0f;
  for (int pi = 0; pi < 16; ++pi) {
    int p = p0 + pi;
    int brow = (p >> 11) << 11;               // batch base point index
    float zc = zs[(size_t)p * OPER + o];
    const int* ip = idx + (size_t)p * 32;
    for (int kk = 0; kk < k; ++kk) {
      int j = ip[kk];
      float h = y1s[(size_t)(brow + j) * OPER + o] + zc;
      sum += h; sumsq += h * h;
    }
  }
  atomicAdd(&stats[s * 256 + o], sum);
  atomicAdd(&stats[s * 256 + 128 + o], sumsq);
}

__global__ void k_edge_coef(const float* __restrict__ stats,
                            const float* __restrict__ g0, const float* __restrict__ g1,
                            const float* __restrict__ g2, const float* __restrict__ be0,
                            const float* __restrict__ be1, const float* __restrict__ be2,
                            float* __restrict__ coef) {
  int t = threadIdx.x;             // 0..383
  int s = t >> 7, o = t & 127;
  float cnt = (float)BN * (float)(8 << s);
  float mu  = stats[s * 256 + o] / cnt;
  float var = stats[s * 256 + 128 + o] / cnt - mu * mu;
  const float* g  = (s == 0) ? g0  : ((s == 1) ? g1  : g2);
  const float* be = (s == 0) ? be0 : ((s == 1) ? be1 : be2);
  float a = g[o] * rsqrtf(var + 1e-5f);
  coef[s * 256 + o]       = a;
  coef[s * 256 + 128 + o] = be[o] - mu * a;
}

// grid (BN, 3), block 128: max over k of leakyrelu(BN(h)) -> featb bf16
__global__ void k_maxpool(const float* __restrict__ y1, const float* __restrict__ z,
                          const int* __restrict__ idx, const float* __restrict__ coef,
                          __bf16* __restrict__ featb) {
  const int s = blockIdx.y;
  const int k = 8 << s;
  const int p = blockIdx.x;
  const int o = threadIdx.x;
  const float* y1s = y1 + (size_t)s * BN * OPER;
  const float* zs  = z  + (size_t)s * BN * OPER;
  int brow = (p >> 11) << 11;
  float a  = coef[s * 256 + o];
  float sh = coef[s * 256 + 128 + o];
  float zc = zs[(size_t)p * OPER + o];
  const int* ip = idx + (size_t)p * 32;
  float m = -3.4e38f;
  for (int kk = 0; kk < k; ++kk) {
    int j = ip[kk];
    float h = y1s[(size_t)(brow + j) * OPER + o] + zc;
    float v = h * a + sh;
    v = (v > 0.0f) ? v : 0.2f * v;
    m = fmaxf(m, v);
  }
  featb[(size_t)p * OUTC + s * OPER + o] = (__bf16)m;
}

// grid (BN/128, OUTC/16), block 256: y = feat * Wf^T + bf  (K = 384)
__global__ void __launch_bounds__(256)
k_final_gemm(const __bf16* __restrict__ featb, const __bf16* __restrict__ wfb,
             const float* __restrict__ bfin, float* __restrict__ ytmp) {
  const int lane = threadIdx.x & 31;
  const int wv   = threadIdx.x >> 5;
  const int row0 = blockIdx.x * 128 + wv * 16;
  const int o0   = blockIdx.y * 16;

  v8f acc = {};
#pragma unroll
  for (int kc = 0; kc < OUTC / 32; ++kc) {
    v16bf a  = load_a_tile(featb, OUTC, row0, kc * 32, lane);
    v16bf bm = load_bt_tile(wfb, OUTC, o0, kc * 32, lane);
    acc = wmma_bf16(a, bm, acc);
  }
  const int   col  = o0 + (lane & 15);
  const float bias = bfin[col];
#pragma unroll
  for (int r = 0; r < 8; ++r) {
    int m = r + ((lane >> 4) << 3);
    ytmp[(size_t)(row0 + m) * OUTC + col] = acc[r] + bias;
  }
}

// grid (BN/128), block 384: channel sums of ytmp
__global__ void k_final_stats(const float* __restrict__ ytmp, float* __restrict__ fstats) {
  int o  = threadIdx.x;            // 0..383
  int r0 = blockIdx.x * 128;
  float sum = 0.0f, sumsq = 0.0f;
  for (int r = 0; r < 128; ++r) {
    float v = ytmp[(size_t)(r0 + r) * OUTC + o];
    sum += v; sumsq += v * v;
  }
  atomicAdd(&fstats[o], sum);
  atomicAdd(&fstats[OUTC + o], sumsq);
}

// grid (PN/256, OUTC, PB), block 256: BN + exact GELU -> out (B, 384, N)
__global__ void k_final_out(const float* __restrict__ ytmp, const float* __restrict__ fstats,
                            const float* __restrict__ gf, const float* __restrict__ bef,
                            float* __restrict__ out) {
  int n = blockIdx.x * blockDim.x + threadIdx.x;
  int o = blockIdx.y;
  int b = blockIdx.z;
  float cnt = (float)BN;
  float mu  = fstats[o] / cnt;
  float var = fstats[OUTC + o] / cnt - mu * mu;
  float a   = gf[o] * rsqrtf(var + 1e-5f);
  float sh  = bef[o] - mu * a;
  float v   = ytmp[(size_t)(b * PN + n) * OUTC + o] * a + sh;
  float ge  = 0.5f * v * (1.0f + erff(v * 0.70710678118654752f));
  out[((size_t)(b * OUTC + o)) * PN + n] = ge;
}

// ---------------------------------------------------------------------------

extern "C" void kernel_launch(void* const* d_in, const int* in_sizes, int n_in,
                              void* d_out, int out_size, void* d_ws, size_t ws_size,
                              hipStream_t stream) {
  const float* x   = (const float*)d_in[0];
  const float* W0  = (const float*)d_in[1];
  const float* b0  = (const float*)d_in[2];
  const float* g0  = (const float*)d_in[3];
  const float* be0 = (const float*)d_in[4];
  const float* W1  = (const float*)d_in[5];
  const float* b1  = (const float*)d_in[6];
  const float* g1  = (const float*)d_in[7];
  const float* be1 = (const float*)d_in[8];
  const float* W2  = (const float*)d_in[9];
  const float* b2  = (const float*)d_in[10];
  const float* g2  = (const float*)d_in[11];
  const float* be2 = (const float*)d_in[12];
  const float* Wf  = (const float*)d_in[13];
  const float* bfv = (const float*)d_in[14];
  const float* gf  = (const float*)d_in[15];
  const float* bef = (const float*)d_in[16];
  float* out = (float*)d_out;

  char* ws = (char*)d_ws;
  size_t off = 0;
  auto alloc = [&](size_t bytes) -> char* {
    char* p = ws + off;
    off += (bytes + 255) & ~(size_t)255;
    return p;
  };

  __bf16* xtb   = (__bf16*)alloc((size_t)BN * PC * 2);          //  4 MiB
  float*  sq    = (float*) alloc((size_t)BN * 4);
  __bf16* w1b   = (__bf16*)alloc((size_t)3 * OPER * PC * 2);
  __bf16* wdb   = (__bf16*)alloc((size_t)3 * OPER * PC * 2);
  __bf16* wfb   = (__bf16*)alloc((size_t)OUTC * OUTC * 2);
  float*  D     = (float*) alloc((size_t)BN * PN * 4);          // 128 MiB
  int*    idx   = (int*)   alloc((size_t)BN * 32 * 4);
  float*  y1    = (float*) alloc((size_t)3 * BN * OPER * 4);    //  24 MiB
  float*  z     = (float*) alloc((size_t)3 * BN * OPER * 4);    //  24 MiB
  float*  stats = (float*) alloc(768 * 4);
  float*  coef  = (float*) alloc(768 * 4);
  __bf16* featb = (__bf16*)alloc((size_t)BN * OUTC * 2);        //  12 MiB
  float*  ytmp  = (float*) alloc((size_t)BN * OUTC * 4);        //  24 MiB
  float*  fstats= (float*) alloc(768 * 4);
  (void)ws_size; (void)in_sizes; (void)n_in; (void)out_size;

  k_zero<<<3, 256, 0, stream>>>(stats, fstats);
  k_prep_w<<<(OUTC * OUTC + 255) / 256, 256, 0, stream>>>(W0, W1, W2, Wf, w1b, wdb, wfb);
  k_transpose<<<BN, PC, 0, stream>>>(x, xtb, sq);
  k_gram<<<dim3(PN / 128, PN / 128, PB), 256, 0, stream>>>(xtb, sq, D);
  k_topk<<<BN / 64, 64, 0, stream>>>(D, idx);
  k_feat_gemm<<<dim3(BN / 128, OPER / 16, 3), 256, 0, stream>>>(xtb, w1b, wdb,
                                                                b0, b1, b2, y1, z);
  k_edge_stats<<<dim3(BN / 16, 3), 128, 0, stream>>>(y1, z, idx, stats);
  k_edge_coef<<<1, 384, 0, stream>>>(stats, g0, g1, g2, be0, be1, be2, coef);
  k_maxpool<<<dim3(BN, 3), 128, 0, stream>>>(y1, z, idx, coef, featb);
  k_final_gemm<<<dim3(BN / 128, OUTC / 16), 256, 0, stream>>>(featb, wfb, bfv, ytmp);
  k_final_stats<<<BN / 128, 384, 0, stream>>>(ytmp, fstats);
  k_final_out<<<dim3(PN / 256, OUTC, PB), 256, 0, stream>>>(ytmp, fstats, gf, bef, out);
}
